// TransformerMoE_27075473834350
// MI455X (gfx1250) — compile-verified, matrix operand order checked by reading, not run on hardware
//
#include <hip/hip_runtime.h>
#include <hip/hip_bf16.h>

// ---------------------------------------------------------------- constants
constexpr int cV  = 32000;
constexpr int cD  = 768;
constexpr int cH  = 12;
constexpr int cHD = 64;
constexpr int cL  = 5;
constexpr int cF  = 2048;
constexpr int cE  = 4;
constexpr int cB  = 8;
constexpr int cS  = 512;
constexpr int cDH = 4 * cD;      // 3072
constexpr int cNT = cB * cS;     // 4096 tokens

typedef _Float16 half_t;
typedef __attribute__((ext_vector_type(16))) _Float16 v16h;
typedef __attribute__((ext_vector_type(8)))  _Float16 h8;
typedef __attribute__((ext_vector_type(8)))  float    v8f;

// XOR-swizzled LDS indexers: rotate the 8-half k-chunk by row bits so the
// transposed scalar stores and the b128 fragment reads spread across banks.
// Swizzle mask only touches k bits >= 3, so 16-byte chunk alignment survives.
__device__ __forceinline__ int sw32(int row, int k) {       // 32-half rows
  return row * 32 + (k ^ ((((row >> 2) ^ (row >> 4)) & 3) << 3));
}
__device__ __forceinline__ int sw64(int row, int k) {       // 64-half rows
  return row * 64 + (k ^ ((((row >> 1) ^ (row >> 3)) & 7) << 3));
}

// ---------------------------------------------------------------- fp32 -> fp16
__global__ __launch_bounds__(256) void cvt_f32_f16(const float* __restrict__ src,
                                                   half_t* __restrict__ dst, size_t n) {
  size_t i = (size_t)blockIdx.x * 256 + threadIdx.x;
  if (i < n) dst[i] = (half_t)src[i];
}

// ---------------------------------------------------------------- embedding
__global__ __launch_bounds__(256) void embed_kernel(const int* __restrict__ ids,
                                                    const float* __restrict__ emb,
                                                    half_t* __restrict__ x) {
  int n = blockIdx.x;
  int id = ids[n];
  id = id < 0 ? 0 : (id >= cV ? cV - 1 : id);
  const float sc = 27.712812921102035f;  // sqrt(768)
  for (int d = threadIdx.x; d < cD; d += 256)
    x[(size_t)n * cD + d] = (half_t)(emb[(size_t)id * cD + d] * sc);
}

// ---------------------------------------------------------------- RoPE (in place on q,k)
__global__ __launch_bounds__(256) void rope_kernel(half_t* __restrict__ q,
                                                   half_t* __restrict__ k) {
  int n = blockIdx.x;
  int pos = n % cS;
  for (int idx = threadIdx.x; idx < cH * 32; idx += 256) {
    int h = idx >> 5, i = idx & 31;
    float invf = __powf(10000.f, -(float)i / 32.f);
    float sn, csn;
    __sincosf((float)pos * invf, &sn, &csn);
    size_t base = (size_t)n * cD + h * cHD;
    float q1 = (float)q[base + i], q2 = (float)q[base + i + 32];
    q[base + i]      = (half_t)(q1 * csn - q2 * sn);
    q[base + i + 32] = (half_t)(q2 * csn + q1 * sn);
    float k1 = (float)k[base + i], k2 = (float)k[base + i + 32];
    k[base + i]      = (half_t)(k1 * csn - k2 * sn);
    k[base + i + 32] = (half_t)(k2 * csn + k1 * sn);
  }
}

// ---------------------------------------------------------------- layernorm
__global__ __launch_bounds__(256) void layernorm_kernel(const half_t* __restrict__ x,
                                                        const float* __restrict__ g,
                                                        const float* __restrict__ b,
                                                        half_t* __restrict__ out) {
  __shared__ float r1[256], r2[256];
  int n = blockIdx.x, t = threadIdx.x;
  const half_t* xr = x + (size_t)n * cD;
  float s = 0.f, s2 = 0.f;
  for (int d = t; d < cD; d += 256) { float v = (float)xr[d]; s += v; s2 += v * v; }
  r1[t] = s; r2[t] = s2;
  __syncthreads();
  for (int st = 128; st > 0; st >>= 1) {
    if (t < st) { r1[t] += r1[t + st]; r2[t] += r2[t + st]; }
    __syncthreads();
  }
  float mu  = r1[0] / cD;
  float var = r2[0] / cD - mu * mu;
  float inv = rsqrtf(var + 1e-5f);
  for (int d = t; d < cD; d += 256)
    out[(size_t)n * cD + d] = (half_t)(((float)xr[d] - mu) * inv * g[d] + b[d]);
}

// ---------------------------------------------------------------- generic WMMA GEMM
// C[M,N] = act(A[M,K] @ W[K,N] + bias[N] (+ resid[M,N]))
// tile 128x64, k-step 32, 256 threads = 8 waves (one 16x64 strip each).
// Double-buffered LDS pipeline; B tile stored transposed+swizzled so both
// fragment operands are read as ds_load_b128 pairs.
template <int ACT, bool RESID, bool OUTF32>
__global__ __launch_bounds__(256) void gemm16(const half_t* __restrict__ A,
                                              const half_t* __restrict__ W,
                                              const float* __restrict__ bias,
                                              const half_t* __restrict__ resid,
                                              void* __restrict__ Cout,
                                              int M, int N, int Kd) {
  __shared__ __align__(16) half_t As[2][128 * 32];
  __shared__ __align__(16) half_t BsT[2][64 * 32];
  const int t = threadIdx.x, wave = t >> 5, lane = t & 31;
  const int m0 = blockIdx.y * 128, n0 = blockIdx.x * 64;
  const int kb = (lane >> 4) * 8;

  // staging coordinates
  const int ar0 = t >> 2,          ac0 = (t & 3) * 8;       // A chunk #t
  const int ar1 = (t + 256) >> 2,  ac1 = ((t + 256) & 3) * 8; // A chunk #t+256
  const int br  = t >> 3,          bc  = (t & 7) * 8;       // W row k0+br, cols n0+bc..+7

  auto loadA0 = [&](int k0) { return *(const h8*)&A[(size_t)(m0 + ar0) * Kd + k0 + ac0]; };
  auto loadA1 = [&](int k0) { return *(const h8*)&A[(size_t)(m0 + ar1) * Kd + k0 + ac1]; };
  auto loadB  = [&](int k0) { return *(const h8*)&W[(size_t)(k0 + br) * N + n0 + bc]; };
  auto stage  = [&](int buf, h8 a0, h8 a1, h8 bvec) {
    *(h8*)&As[buf][sw32(ar0, ac0)] = a0;
    *(h8*)&As[buf][sw32(ar1, ac1)] = a1;
#pragma unroll
    for (int i = 0; i < 8; ++i) BsT[buf][sw32(bc + i, br)] = bvec[i];  // transpose
  };

  v8f acc[4] = {};
  const int nIter = Kd / 32;

  stage(0, loadA0(0), loadA1(0), loadB(0));
  __syncthreads();

  for (int it = 0; it < nIter; ++it) {
    const int cur = it & 1;
    h8 ra0, ra1, rb;
    if (it + 1 < nIter) {
      const int k0 = (it + 1) * 32;
      ra0 = loadA0(k0); ra1 = loadA1(k0); rb = loadB(k0);
      if (it + 4 < nIter)   // far-ahead prefetch of the weight stream
        __builtin_prefetch(&W[(size_t)((it + 4) * 32 + br) * N + n0 + bc], 0, 1);
    }

    v16h a;
    {
      const int row = wave * 16 + (lane & 15);
      const half_t* ap0 = &As[cur][sw32(row, kb)];
      const half_t* ap1 = &As[cur][sw32(row, kb + 16)];
#pragma unroll
      for (int j = 0; j < 8; ++j) { a[j] = ap0[j]; a[j + 8] = ap1[j]; }
    }
#pragma unroll
    for (int c = 0; c < 4; ++c) {
      const int bn = c * 16 + (lane & 15);
      const half_t* bp0 = &BsT[cur][sw32(bn, kb)];
      const half_t* bp1 = &BsT[cur][sw32(bn, kb + 16)];
      v16h bf;
#pragma unroll
      for (int j = 0; j < 8; ++j) { bf[j] = bp0[j]; bf[j + 8] = bp1[j]; }
      acc[c] = __builtin_amdgcn_wmma_f32_16x16x32_f16(false, a, false, bf,
                                                      (short)0, acc[c], false, false);
    }

    if (it + 1 < nIter) {
      stage(cur ^ 1, ra0, ra1, rb);
      __syncthreads();
    }
  }

#pragma unroll
  for (int c = 0; c < 4; ++c) {
#pragma unroll
    for (int v = 0; v < 8; ++v) {
      int row = m0 + wave * 16 + (lane >> 4) * 8 + v;
      int col = n0 + c * 16 + (lane & 15);
      float val = acc[c][v] + bias[col];
      if constexpr (RESID) val += (float)resid[(size_t)row * N + col];
      if constexpr (ACT == 1) val = fmaxf(val, 0.f);
      if constexpr (OUTF32) ((float*)Cout)[(size_t)row * N + col] = val;
      else ((half_t*)Cout)[(size_t)row * N + col] = (half_t)val;
    }
  }
}

// ---------------------------------------------------------------- flash attention
// one block per (b*h, 128-query tile); 8 waves, wave owns 16 query rows.
// V tile stored transposed+swizzled so the P@V B-fragment reads are b128.
__global__ __launch_bounds__(256) void attention_kernel(const half_t* __restrict__ q,
                                                        const half_t* __restrict__ k,
                                                        const half_t* __restrict__ v,
                                                        half_t* __restrict__ ctx) {
  __shared__ __align__(16) half_t Qs[128 * 64];
  __shared__ __align__(16) half_t Ks[64 * 64];
  __shared__ __align__(16) half_t VsT[64 * 64];   // [hd][kv]
  __shared__ __align__(16) half_t Ps[8][16 * 64];

  const int bh = blockIdx.x;
  const int b = bh / cH, h = bh % cH;
  const int q0 = blockIdx.y * 128;
  const int t = threadIdx.x, wave = t >> 5, lane = t & 31;
  const int kb = (lane >> 4) * 8;
  const int qrow = wave * 16;

#pragma unroll
  for (int i = 0; i < 4; ++i) {                  // Q tile 128x64
    int ch = t + i * 256;
    int r = ch >> 3, c8 = (ch & 7) * 8;
    *(h8*)&Qs[sw64(r, c8)] =
        *(const h8*)&q[(size_t)(b * cS + q0 + r) * cD + h * cHD + c8];
  }

  v8f acc[4] = {};
  float mrow[8], lrow[8];
#pragma unroll
  for (int vv = 0; vv < 8; ++vv) { mrow[vv] = -1e30f; lrow[vv] = 0.f; }

  for (int j0 = 0; j0 < cS; j0 += 64) {
    __syncthreads();
#pragma unroll
    for (int i = 0; i < 2; ++i) {                // K,V tiles 64x64
      int ch = t + i * 256;
      int r = ch >> 3, c8 = (ch & 7) * 8;
      *(h8*)&Ks[sw64(r, c8)] =
          *(const h8*)&k[(size_t)(b * cS + j0 + r) * cD + h * cHD + c8];
      h8 vv8 = *(const h8*)&v[(size_t)(b * cS + j0 + r) * cD + h * cHD + c8];
#pragma unroll
      for (int j = 0; j < 8; ++j) VsT[sw64(c8 + j, r)] = vv8[j];  // transpose
    }
    __syncthreads();

    // S = Q @ K^T  (contraction over HD=64 -> 2 wmma k-steps)
    v8f s[4] = {};
#pragma unroll
    for (int ks = 0; ks < 64; ks += 32) {
      v16h a;
      {
        const int row = qrow + (lane & 15);
        const half_t* ap0 = &Qs[sw64(row, ks + kb)];
        const half_t* ap1 = &Qs[sw64(row, ks + kb + 16)];
#pragma unroll
        for (int j = 0; j < 8; ++j) { a[j] = ap0[j]; a[j + 8] = ap1[j]; }
      }
#pragma unroll
      for (int c = 0; c < 4; ++c) {
        const int n = c * 16 + (lane & 15);
        const half_t* bp0 = &Ks[sw64(n, ks + kb)];
        const half_t* bp1 = &Ks[sw64(n, ks + kb + 16)];
        v16h bf;
#pragma unroll
        for (int j = 0; j < 8; ++j) { bf[j] = bp0[j]; bf[j + 8] = bp1[j]; }
        s[c] = __builtin_amdgcn_wmma_f32_16x16x32_f16(false, a, false, bf,
                                                      (short)0, s[c], false, false);
      }
    }

    // online softmax update (rows replicated across 16-lane groups)
#pragma unroll
    for (int vv = 0; vv < 8; ++vv) {
      float lm = -1e30f;
#pragma unroll
      for (int c = 0; c < 4; ++c) { s[c][vv] *= 0.125f; lm = fmaxf(lm, s[c][vv]); }
      for (int mk = 1; mk < 16; mk <<= 1) lm = fmaxf(lm, __shfl_xor(lm, mk, 32));
      float mnew = fmaxf(mrow[vv], lm);
      float scal = __expf(mrow[vv] - mnew);
      mrow[vv] = mnew;
      float rs = 0.f;
#pragma unroll
      for (int c = 0; c < 4; ++c) {
        float p = __expf(s[c][vv] - mnew);
        s[c][vv] = p;
        rs += p;
      }
      for (int mk = 1; mk < 16; mk <<= 1) rs += __shfl_xor(rs, mk, 32);
      lrow[vv] = lrow[vv] * scal + rs;
#pragma unroll
      for (int c = 0; c < 4; ++c) acc[c][vv] *= scal;
    }

    // stage P (C-layout -> A-layout) through per-wave LDS patch
#pragma unroll
    for (int c = 0; c < 4; ++c)
#pragma unroll
      for (int vv = 0; vv < 8; ++vv)
        Ps[wave][sw64((lane >> 4) * 8 + vv, c * 16 + (lane & 15))] = (half_t)s[c][vv];

    // ctx += P @ V  (contraction over 64 keys -> 2 wmma k-steps)
#pragma unroll
    for (int ks = 0; ks < 64; ks += 32) {
      v16h a;
      {
        const half_t* ap0 = &Ps[wave][sw64(lane & 15, ks + kb)];
        const half_t* ap1 = &Ps[wave][sw64(lane & 15, ks + kb + 16)];
#pragma unroll
        for (int j = 0; j < 8; ++j) { a[j] = ap0[j]; a[j + 8] = ap1[j]; }
      }
#pragma unroll
      for (int c = 0; c < 4; ++c) {
        const int n = c * 16 + (lane & 15);
        const half_t* bp0 = &VsT[sw64(n, ks + kb)];
        const half_t* bp1 = &VsT[sw64(n, ks + kb + 16)];
        v16h bf;
#pragma unroll
        for (int j = 0; j < 8; ++j) { bf[j] = bp0[j]; bf[j + 8] = bp1[j]; }
        acc[c] = __builtin_amdgcn_wmma_f32_16x16x32_f16(false, a, false, bf,
                                                        (short)0, acc[c], false, false);
      }
    }
  }

#pragma unroll
  for (int c = 0; c < 4; ++c)
#pragma unroll
    for (int vv = 0; vv < 8; ++vv) {
      int r = q0 + wave * 16 + (lane >> 4) * 8 + vv;
      int col = c * 16 + (lane & 15);
      ctx[(size_t)(b * cS + r) * cD + h * cHD + col] = (half_t)(acc[c][vv] / lrow[vv]);
    }
}

// ---------------------------------------------------------------- MoE gate (top-2)
__global__ __launch_bounds__(128) void gate_kernel(const half_t* __restrict__ x,
                                                   const float* __restrict__ gw,
                                                   const float* __restrict__ gb,
                                                   int* __restrict__ idxo,
                                                   float* __restrict__ wto) {
  __shared__ float red[128][cE];
  int n = blockIdx.x, t = threadIdx.x;
  float a[cE] = {0.f, 0.f, 0.f, 0.f};
  for (int d = t; d < cD; d += 128) {
    float xv = (float)x[(size_t)n * cD + d];
#pragma unroll
    for (int e = 0; e < cE; ++e) a[e] += xv * gw[d * cE + e];
  }
#pragma unroll
  for (int e = 0; e < cE; ++e) red[t][e] = a[e];
  __syncthreads();
  for (int st = 64; st > 0; st >>= 1) {
    if (t < st)
#pragma unroll
      for (int e = 0; e < cE; ++e) red[t][e] += red[t + st][e];
    __syncthreads();
  }
  if (t == 0) {
    float lg[cE];
#pragma unroll
    for (int e = 0; e < cE; ++e) lg[e] = red[0][e] + gb[e];
    int i0 = 0;
    for (int e = 1; e < cE; ++e) if (lg[e] > lg[i0]) i0 = e;
    int i1 = -1;
    for (int e = 0; e < cE; ++e)
      if (e != i0 && (i1 < 0 || lg[e] > lg[i1])) i1 = e;
    float e0 = 1.f, e1 = __expf(lg[i1] - lg[i0]);
    float inv = 1.f / (e0 + e1);
    idxo[n * 2] = i0; idxo[n * 2 + 1] = i1;
    wto[n * 2] = e0 * inv; wto[n * 2 + 1] = e1 * inv;
  }
}

// ---------------------------------------------------------------- MoE combine
__global__ __launch_bounds__(256) void moe_combine_kernel(const half_t* __restrict__ eouts,
                                                          const int* __restrict__ idx,
                                                          const float* __restrict__ wt,
                                                          half_t* __restrict__ out) {
  int n = blockIdx.x, t = threadIdx.x;
  int i0 = idx[n * 2], i1 = idx[n * 2 + 1];
  float w0 = wt[n * 2], w1 = wt[n * 2 + 1];
  for (int d = t; d < cD; d += 256)
    out[(size_t)n * cD + d] =
        (half_t)(w0 * (float)eouts[((size_t)i0 * cNT + n) * cD + d] +
                 w1 * (float)eouts[((size_t)i1 * cNT + n) * cD + d]);
}

// ================================================================ launch
extern "C" void kernel_launch(void* const* d_in, const int* in_sizes, int n_in,
                              void* d_out, int out_size, void* d_ws, size_t ws_size,
                              hipStream_t stream) {
  (void)in_sizes; (void)n_in; (void)out_size; (void)ws_size;

  const int*   ids  = (const int*)d_in[0];
  const float* emb  = (const float*)d_in[1];
  auto LP = [&](int l, int j) { return (const float*)d_in[2 + l * 16 + j]; };
  const float* gw   = (const float*)d_in[82];
  const float* gb   = (const float*)d_in[83];
  const float* e1w  = (const float*)d_in[84];
  const float* e1b  = (const float*)d_in[85];
  const float* e2w  = (const float*)d_in[86];
  const float* e2b  = (const float*)d_in[87];
  const float* lng  = (const float*)d_in[88];
  const float* lnb  = (const float*)d_in[89];
  const float* outw = (const float*)d_in[90];
  const float* outb = (const float*)d_in[91];

  // ---- workspace carve (256B aligned)
  char* base = (char*)d_ws;
  size_t off = 0;
  auto ah = [&](size_t n) { half_t* p = (half_t*)(base + off);
                            off += ((n * sizeof(half_t) + 255) / 256) * 256; return p; };
  auto af = [&](size_t n) { float* p = (float*)(base + off);
                            off += ((n * sizeof(float) + 255) / 256) * 256; return p; };
  auto ai = [&](size_t n) { int* p = (int*)(base + off);
                            off += ((n * sizeof(int) + 255) / 256) * 256; return p; };

  half_t *wq[cL], *wk[cL], *wv[cL], *wo[cL], *wf1[cL], *wf2[cL];
  for (int l = 0; l < cL; ++l) {
    wq[l]  = ah((size_t)cD * cD);
    wk[l]  = ah((size_t)cD * cD);
    wv[l]  = ah((size_t)cD * cD);
    wo[l]  = ah((size_t)cD * cD);
    wf1[l] = ah((size_t)cD * cF);
    wf2[l] = ah((size_t)cF * cD);
  }
  half_t* we1  = ah((size_t)cE * cD * cDH);
  half_t* we2  = ah((size_t)cE * cDH * cD);
  half_t* wout = ah((size_t)cD * cV);

  half_t* bufA = ah((size_t)cNT * cD);   // residual stream
  half_t* bufB = ah((size_t)cNT * cD);   // pre-LN scratch
  half_t* bufQ = ah((size_t)cNT * cD);   // q / eouts[0]
  half_t* bufK = ah((size_t)cNT * cD);   // k / eouts[1]
  half_t* bufV = ah((size_t)cNT * cD);   // v / eouts[2]
  half_t* bufC = ah((size_t)cNT * cD);   // ctx / eouts[3]
  half_t* bufH = ah((size_t)cNT * cDH);  // ffn / expert hidden
  half_t* eouts = bufQ;                  // contiguous 4 x [NT,D]
  int*    gidx = ai((size_t)cNT * 2);
  float*  gwt  = af((size_t)cNT * 2);

  auto cvt = [&](const float* s, half_t* d, size_t n) {
    cvt_f32_f16<<<dim3((unsigned)((n + 255) / 256)), 256, 0, stream>>>(s, d, n);
  };

  // ---- one-time weight conversion to f16
  for (int l = 0; l < cL; ++l) {
    cvt(LP(l, 0),  wq[l],  (size_t)cD * cD);
    cvt(LP(l, 2),  wk[l],  (size_t)cD * cD);
    cvt(LP(l, 4),  wv[l],  (size_t)cD * cD);
    cvt(LP(l, 6),  wo[l],  (size_t)cD * cD);
    cvt(LP(l, 10), wf1[l], (size_t)cD * cF);
    cvt(LP(l, 12), wf2[l], (size_t)cF * cD);
  }
  cvt(e1w,  we1,  (size_t)cE * cD * cDH);
  cvt(e2w,  we2,  (size_t)cE * cDH * cD);
  cvt(outw, wout, (size_t)cD * cV);

  // ---- embedding
  embed_kernel<<<cNT, 256, 0, stream>>>(ids, emb, bufA);

  dim3 blk(256);
  dim3 gD(cD / 64, cNT / 128);
  dim3 gF(cF / 64, cNT / 128);
  dim3 gDH(cDH / 64, cNT / 128);
  dim3 gV(cV / 64, cNT / 128);

  // ---- encoder layers
  for (int l = 0; l < cL; ++l) {
    gemm16<0, false, false><<<gD, blk, 0, stream>>>(bufA, wq[l], LP(l, 1), nullptr, bufQ, cNT, cD, cD);
    gemm16<0, false, false><<<gD, blk, 0, stream>>>(bufA, wk[l], LP(l, 3), nullptr, bufK, cNT, cD, cD);
    gemm16<0, false, false><<<gD, blk, 0, stream>>>(bufA, wv[l], LP(l, 5), nullptr, bufV, cNT, cD, cD);
    rope_kernel<<<cNT, 256, 0, stream>>>(bufQ, bufK);
    attention_kernel<<<dim3(cB * cH, cS / 128), blk, 0, stream>>>(bufQ, bufK, bufV, bufC);
    gemm16<0, true, false><<<gD, blk, 0, stream>>>(bufC, wo[l], LP(l, 7), bufA, bufB, cNT, cD, cD);
    layernorm_kernel<<<cNT, 256, 0, stream>>>(bufB, LP(l, 8), LP(l, 9), bufA);
    gemm16<1, false, false><<<gF, blk, 0, stream>>>(bufA, wf1[l], LP(l, 11), nullptr, bufH, cNT, cF, cD);
    gemm16<0, true, false><<<gD, blk, 0, stream>>>(bufH, wf2[l], LP(l, 13), bufA, bufB, cNT, cD, cF);
    layernorm_kernel<<<cNT, 256, 0, stream>>>(bufB, LP(l, 14), LP(l, 15), bufA);
  }

  // ---- MoE (dense over all experts, matching reference math)
  gate_kernel<<<cNT, 128, 0, stream>>>(bufA, gw, gb, gidx, gwt);
  for (int e = 0; e < cE; ++e) {
    gemm16<1, false, false><<<gDH, blk, 0, stream>>>(
        bufA, we1 + (size_t)e * cD * cDH, e1b + (size_t)e * cDH, nullptr, bufH, cNT, cDH, cD);
    gemm16<0, false, false><<<gD, blk, 0, stream>>>(
        bufH, we2 + (size_t)e * cDH * cD, e2b + (size_t)e * cD, nullptr,
        eouts + (size_t)e * cNT * cD, cNT, cD, cDH);
  }
  moe_combine_kernel<<<cNT, 256, 0, stream>>>(eouts, gidx, gwt, bufB);

  // ---- final LN + vocab projection (fp32 out)
  layernorm_kernel<<<cNT, 256, 0, stream>>>(bufB, lng, lnb, bufA);
  gemm16<0, false, true><<<gV, blk, 0, stream>>>(bufA, wout, outb, nullptr, d_out, cNT, cV, cD);
}
